// AttentionLayer_10015863734274
// MI455X (gfx1250) — matrix operand order repaired and checked by reading, not compiled, not run on hardware
//
#include <hip/hip_runtime.h>
#include <hip/hip_bf16.h>

// ---------------------------------------------------------------------------
// Problem constants (from the reference)
// ---------------------------------------------------------------------------
#define EMBED 1024
#define DH    128          // KQ == VDIM == 128
#define SEQ   2048
#define BATCH 8
#define ROWS  (BATCH * SEQ)   // 16384

// ---------------------------------------------------------------------------
// WMMA types / helpers (CDNA5 wave32, v_wmma_f32_16x16x32_bf16)
// ---------------------------------------------------------------------------
typedef __attribute__((ext_vector_type(16))) __bf16 bf16x16;
typedef __attribute__((ext_vector_type(8)))  __bf16 bf16x8;
typedef __attribute__((ext_vector_type(8)))  float  f32x8;

static __device__ __forceinline__ bf16x16 combine8(bf16x8 lo, bf16x8 hi) {
    union { bf16x16 v; bf16x8 h[2]; } u;
    u.h[0] = lo; u.h[1] = hi;
    return u.v;
}

static __device__ __forceinline__ f32x8 wmma_bf16(bf16x16 a, bf16x16 b, f32x8 c) {
    // 8 args: (neg_a, A, neg_b, B, c_mod, C, reuse_a, reuse_b)
    return __builtin_amdgcn_wmma_f32_16x16x32_bf16(false, a, false, b,
                                                   (short)0, c, false, false);
}

// pack float4 -> 4 bf16 (8 bytes)
static __device__ __forceinline__ void store_bf4(__bf16* dst, float4 f) {
    __bf16 t[4] = { (__bf16)f.x, (__bf16)f.y, (__bf16)f.z, (__bf16)f.w };
    *(uint2*)dst = *(const uint2*)t;
}

// async global->LDS 16B-per-lane copy (CDNA5, ASYNCcnt tracked)
static __device__ __forceinline__ void async_copy_b128(unsigned lds_addr,
                                                       const void* gaddr) {
    unsigned long long ga = (unsigned long long)(uintptr_t)gaddr;
    asm volatile("global_load_async_to_lds_b128 %0, %1, off"
                 :: "v"(lds_addr), "v"(ga) : "memory");
}
static __device__ __forceinline__ void wait_asynccnt0() {
    asm volatile("s_wait_asynccnt 0x0" ::: "memory");
}

// ---------------------------------------------------------------------------
// Kernel 1: projection  out_bf16[M,128] = bf16( (x[M,E] @ W[E,128] + b)*scl )
// Block: 256 threads (8 waves). Tile 128(M) x 128(N). K chunks of 32,
// double-buffered LDS so global loads overlap the WMMAs.
// ---------------------------------------------------------------------------
__global__ __launch_bounds__(256) void qkv_proj_kernel(
    const float* __restrict__ x, const float* __restrict__ W,
    const float* __restrict__ bias, __bf16* __restrict__ outbf,
    float outScale)
{
    __shared__ __align__(16) __bf16 sA [2][128 * 32];   // x tile   [m][k]
    __shared__ __align__(32) __bf16 sBt[2][128 * 32];   // W^T tile [n][k]

    const int tid  = threadIdx.x;
    const int wave = tid >> 5;
    const int lane = tid & 31;
    const int l16  = lane & 15;
    const int hl   = lane >> 4;
    const int row0 = blockIdx.x * 128;

    // per-thread staging geometry (4 float4 each for x and W per chunk)
    int xr[4], xk[4], wk[4], wn[4];
#pragma unroll
    for (int j = 0; j < 4; ++j) {
        int f4 = tid + j * 256;
        xr[j] = f4 >> 3;  xk[j] = (f4 & 7) * 4;    // x: [128 m][32 k]
        wk[j] = f4 >> 5;  wn[j] = (f4 & 31) * 4;   // W chunk: [32 k][128 n]
    }

    float4 xa[4], wa[4];
    auto load_tiles = [&](int kb) {
#pragma unroll
        for (int j = 0; j < 4; ++j)
            xa[j] = *(const float4*)&x[(size_t)(row0 + xr[j]) * EMBED + kb + xk[j]];
#pragma unroll
        for (int j = 0; j < 4; ++j)
            wa[j] = *(const float4*)&W[(size_t)(kb + wk[j]) * DH + wn[j]];
    };
    auto store_tiles = [&](int buf) {
#pragma unroll
        for (int j = 0; j < 4; ++j)
            store_bf4(&sA[buf][xr[j] * 32 + xk[j]], xa[j]);
#pragma unroll
        for (int j = 0; j < 4; ++j) {           // transposed: sBt[n][k]
            sBt[buf][(wn[j] + 0) * 32 + wk[j]] = (__bf16)wa[j].x;
            sBt[buf][(wn[j] + 1) * 32 + wk[j]] = (__bf16)wa[j].y;
            sBt[buf][(wn[j] + 2) * 32 + wk[j]] = (__bf16)wa[j].z;
            sBt[buf][(wn[j] + 3) * 32 + wk[j]] = (__bf16)wa[j].w;
        }
    };

    f32x8 acc[8];
#pragma unroll
    for (int t = 0; t < 8; ++t) {
        float bv = bias[t * 16 + l16];
#pragma unroll
        for (int r = 0; r < 8; ++r) acc[t][r] = bv;
    }

    load_tiles(0);
    store_tiles(0);
    __syncthreads();

    for (int kb = 0; kb < EMBED; kb += 32) {
        const int  cur  = (kb >> 5) & 1;
        const bool more = (kb + 32) < EMBED;
        if (more) load_tiles(kb + 32);          // global loads in flight

        const __bf16* ap = &sA[cur][(wave * 16 + l16) * 32];
        bf16x16 a = combine8(*(const bf16x8*)(ap + 16 * hl),
                             *(const bf16x8*)(ap + 16 * hl + 8));
#pragma unroll
        for (int t = 0; t < 8; ++t) {
            bf16x16 b = *(const bf16x16*)&sBt[cur][(t * 16 + l16) * 32 + 16 * hl];
            acc[t] = wmma_bf16(a, b, acc[t]);
        }
        if (more) store_tiles(cur ^ 1);         // fill the other buffer
        __syncthreads();
    }

    // C layout: element r -> row (r + 8*hl), col (t*16 + l16)
#pragma unroll
    for (int t = 0; t < 8; ++t)
#pragma unroll
        for (int r = 0; r < 8; ++r) {
            int row = row0 + wave * 16 + r + 8 * hl;
            int col = t * 16 + l16;
            outbf[(size_t)row * DH + col] = (__bf16)(acc[t][r] * outScale);
        }
}

// ---------------------------------------------------------------------------
// Kernel 2: flash attention. Block: 128 threads (4 waves), 64 query rows,
// one batch. Key blocks of 64, d = 128, ping-pong double buffering:
// next K block staged with CDNA5 async global->LDS copies and next V block
// loaded into registers while the current block's WMMAs run. Q pre-scaled
// by 1/sqrt(d) at projection time.
// ---------------------------------------------------------------------------
__global__ __launch_bounds__(128) void attn_kernel(
    const __bf16* __restrict__ Q, const __bf16* __restrict__ K,
    const __bf16* __restrict__ V, __bf16* __restrict__ H)
{
    __shared__ __align__(32) __bf16 sK [2][64 * 128];    // [key][d]
    __shared__ __align__(32) __bf16 sVt[2][128 * 64];    // [n][key]
    __shared__ __align__(32) __bf16 sP [4 * 16 * 64];    // per-wave P tile

    const int tid  = threadIdx.x;
    const int wave = tid >> 5;
    const int lane = tid & 31;
    const int l16  = lane & 15;
    const int hl   = lane >> 4;
    const int b    = blockIdx.y;
    const int q0   = blockIdx.x * 64;
    const size_t baseB = (size_t)b * SEQ * DH;

    const unsigned sK_base0 = (unsigned)(uintptr_t)&sK[0][0];  // LDS byte offs
    const unsigned sK_base1 = (unsigned)(uintptr_t)&sK[1][0];

    // V staging geometry: 2 adjacent keys per thread -> packed b32 LDS stores
    int vn4[8], vkey[8];
#pragma unroll
    for (int j = 0; j < 8; ++j) {
        int idx = tid + j * 128;            // (key-pair, n-group-of-4)
        vn4[j]  = (idx & 31) * 4;
        vkey[j] = (idx >> 5) * 2;
    }
    uint2 vreg[8][2];

    auto issue_asyncK = [&](int k0, int buf) {
        const char* g = (const char*)(K + baseB + (size_t)k0 * DH);
        unsigned base = buf ? sK_base1 : sK_base0;
#pragma unroll
        for (int i = 0; i < 8; ++i)
            async_copy_b128(base + tid * 16 + i * 2048,
                            g + tid * 16 + i * 2048);
    };
    auto load_V = [&](int k0) {
        const __bf16* gV = V + baseB + (size_t)k0 * DH;
#pragma unroll
        for (int j = 0; j < 8; ++j) {
            vreg[j][0] = *(const uint2*)&gV[(size_t)(vkey[j] + 0) * DH + vn4[j]];
            vreg[j][1] = *(const uint2*)&gV[(size_t)(vkey[j] + 1) * DH + vn4[j]];
        }
    };
    auto store_V = [&](int buf) {
#pragma unroll
        for (int j = 0; j < 8; ++j) {
            const __bf16* e0 = (const __bf16*)&vreg[j][0];
            const __bf16* e1 = (const __bf16*)&vreg[j][1];
#pragma unroll
            for (int i = 0; i < 4; ++i) {
                __bf16 t[2] = { e0[i], e1[i] };
                *(unsigned*)&sVt[buf][(vn4[j] + i) * 64 + vkey[j]] =
                    *(const unsigned*)t;
            }
        }
    };

    // Q fragments in registers: 4 chunks of K=32 over d=128
    bf16x16 aq[4];
    {
        const __bf16* qp = Q + baseB + (size_t)(q0 + wave * 16 + l16) * DH;
#pragma unroll
        for (int c = 0; c < 4; ++c)
            aq[c] = combine8(*(const bf16x8*)(qp + c * 32 + 16 * hl),
                             *(const bf16x8*)(qp + c * 32 + 16 * hl + 8));
    }

    f32x8 o[8];
#pragma unroll
    for (int t = 0; t < 8; ++t)
#pragma unroll
        for (int r = 0; r < 8; ++r) o[t][r] = 0.0f;
    float m_i[8], l_i[8];
#pragma unroll
    for (int r = 0; r < 8; ++r) { m_i[r] = -1e30f; l_i[r] = 0.0f; }

    // prologue: stage block 0 into buffer 0
    issue_asyncK(0, 0);
    load_V(0);
    store_V(0);
    wait_asynccnt0();
    __syncthreads();

    for (int k0 = 0; k0 < SEQ; k0 += 64) {
        const int  cur  = (k0 >> 6) & 1;
        const bool more = (k0 + 64) < SEQ;

        // kick off next block's staging: async K -> other LDS buffer,
        // V -> registers. Both overlap this block's compute.
        if (more) {
            issue_asyncK(k0 + 64, cur ^ 1);
            load_V(k0 + 64);
        }
        if (k0 + 128 < SEQ) {   // keep L2 one more block ahead
            __builtin_prefetch((const char*)(K + baseB + (size_t)(k0 + 128) * DH) + tid * 128);
            __builtin_prefetch((const char*)(V + baseB + (size_t)(k0 + 128) * DH) + tid * 128);
        }

        // ---- S = Q K^T  (16 x 64 per wave; Q pre-scaled) ----
        f32x8 s[4];
#pragma unroll
        for (int t = 0; t < 4; ++t)
#pragma unroll
            for (int r = 0; r < 8; ++r) s[t][r] = 0.0f;
#pragma unroll
        for (int t = 0; t < 4; ++t) {
#pragma unroll
            for (int c = 0; c < 4; ++c) {
                bf16x16 bf = *(const bf16x16*)
                    &sK[cur][(t * 16 + l16) * 128 + c * 32 + 16 * hl];
                s[t] = wmma_bf16(aq[c], bf, s[t]);
            }
        }

        // park next V block in the other LDS buffer (frees the registers;
        // all reads of buffer cur^1 finished at the previous end barrier)
        if (more) store_V(cur ^ 1);

        // ---- online softmax (rows r + 8*hl, cols across 16 lanes) ----
#pragma unroll
        for (int r = 0; r < 8; ++r) {
            float v = fmaxf(fmaxf(s[0][r], s[1][r]), fmaxf(s[2][r], s[3][r]));
#pragma unroll
            for (int off = 8; off >= 1; off >>= 1)
                v = fmaxf(v, __shfl_xor(v, off, 32));
            float newm = fmaxf(m_i[r], v);
            float corr = __expf(m_i[r] - newm);
            float sum = 0.0f;
#pragma unroll
            for (int t = 0; t < 4; ++t) {
                float pv = __expf(s[t][r] - newm);
                s[t][r] = pv;
                sum += pv;
            }
#pragma unroll
            for (int off = 8; off >= 1; off >>= 1)
                sum += __shfl_xor(sum, off, 32);
            l_i[r] = l_i[r] * corr + sum;
            m_i[r] = newm;
#pragma unroll
            for (int t = 0; t < 8; ++t) o[t][r] *= corr;
        }

        // ---- write P tile (C layout -> row-major in per-wave LDS) ----
        {
            __bf16* Pw = sP + wave * 16 * 64;
#pragma unroll
            for (int t = 0; t < 4; ++t)
#pragma unroll
                for (int r = 0; r < 8; ++r)
                    Pw[(r + 8 * hl) * 64 + t * 16 + l16] = (__bf16)s[t][r];
        }
        __syncthreads();                       // P visible to both halves

        // ---- O += P @ V ----
#pragma unroll
        for (int kc = 0; kc < 2; ++kc) {
            const __bf16* pp = sP + wave * 16 * 64 + l16 * 64 + kc * 32;
            bf16x16 af = combine8(*(const bf16x8*)(pp + 16 * hl),
                                  *(const bf16x8*)(pp + 16 * hl + 8));
#pragma unroll
            for (int t = 0; t < 8; ++t) {
                bf16x16 bf = *(const bf16x16*)
                    &sVt[cur][(t * 16 + l16) * 64 + kc * 32 + 16 * hl];
                o[t] = wmma_bf16(af, bf, o[t]);
            }
        }

        wait_asynccnt0();                      // next K block landed in LDS
        __syncthreads();                       // everyone done with buffers
    }

    // ---- epilogue: O / l -> bf16 H ----
#pragma unroll
    for (int r = 0; r < 8; ++r) {
        float inv = 1.0f / l_i[r];
        int row = q0 + wave * 16 + r + 8 * hl;
#pragma unroll
        for (int t = 0; t < 8; ++t)
            H[baseB + (size_t)row * DH + t * 16 + l16] = (__bf16)(o[t][r] * inv);
    }
}

// ---------------------------------------------------------------------------
// Kernel 3: out[M,E] = H[M,128] @ Wo[128,E] + bo   (fp32 out)
// Block: 256 threads (8 waves). Tile 128(M) x 128(N). K = 128, double-buffered.
// ---------------------------------------------------------------------------
__global__ __launch_bounds__(256) void out_proj_kernel(
    const __bf16* __restrict__ Hm, const float* __restrict__ Wo,
    const float* __restrict__ bo, float* __restrict__ out)
{
    __shared__ __align__(16) __bf16 sA [2][128 * 32];
    __shared__ __align__(32) __bf16 sBt[2][128 * 32];

    const int tid  = threadIdx.x;
    const int wave = tid >> 5;
    const int lane = tid & 31;
    const int l16  = lane & 15;
    const int hl   = lane >> 4;
    const int row0 = blockIdx.x * 128;
    const int col0 = blockIdx.y * 128;

    // A staging: 4096 bf16 = 512 uint4 -> 2 per thread
    int ar[2], ak[2];
#pragma unroll
    for (int j = 0; j < 2; ++j) {
        int eb = (tid + j * 256) * 8;
        ar[j] = eb >> 5;  ak[j] = eb & 31;
    }
    int wk[4], wn[4];
#pragma unroll
    for (int j = 0; j < 4; ++j) {
        int f4 = tid + j * 256;
        wk[j] = f4 >> 5;  wn[j] = (f4 & 31) * 4;
    }

    uint4  ha[2];
    float4 wa[4];
    auto load_tiles = [&](int kb) {
#pragma unroll
        for (int j = 0; j < 2; ++j)
            ha[j] = *(const uint4*)&Hm[(size_t)(row0 + ar[j]) * DH + kb + ak[j]];
#pragma unroll
        for (int j = 0; j < 4; ++j)
            wa[j] = *(const float4*)&Wo[(size_t)(kb + wk[j]) * EMBED + col0 + wn[j]];
    };
    auto store_tiles = [&](int buf) {
#pragma unroll
        for (int j = 0; j < 2; ++j)
            *(uint4*)&sA[buf][ar[j] * 32 + ak[j]] = ha[j];
#pragma unroll
        for (int j = 0; j < 4; ++j) {
            sBt[buf][(wn[j] + 0) * 32 + wk[j]] = (__bf16)wa[j].x;
            sBt[buf][(wn[j] + 1) * 32 + wk[j]] = (__bf16)wa[j].y;
            sBt[buf][(wn[j] + 2) * 32 + wk[j]] = (__bf16)wa[j].z;
            sBt[buf][(wn[j] + 3) * 32 + wk[j]] = (__bf16)wa[j].w;
        }
    };

    f32x8 acc[8];
#pragma unroll
    for (int t = 0; t < 8; ++t) {
        float bv = bo[col0 + t * 16 + l16];
#pragma unroll
        for (int r = 0; r < 8; ++r) acc[t][r] = bv;
    }

    load_tiles(0);
    store_tiles(0);
    __syncthreads();

    for (int kb = 0; kb < DH; kb += 32) {
        const int  cur  = (kb >> 5) & 1;
        const bool more = (kb + 32) < DH;
        if (more) load_tiles(kb + 32);

        const __bf16* ap = &sA[cur][(wave * 16 + l16) * 32];
        bf16x16 a = combine8(*(const bf16x8*)(ap + 16 * hl),
                             *(const bf16x8*)(ap + 16 * hl + 8));
#pragma unroll
        for (int t = 0; t < 8; ++t) {
            bf16x16 b = *(const bf16x16*)&sBt[cur][(t * 16 + l16) * 32 + 16 * hl];
            acc[t] = wmma_bf16(a, b, acc[t]);
        }
        if (more) store_tiles(cur ^ 1);
        __syncthreads();
    }

#pragma unroll
    for (int t = 0; t < 8; ++t)
#pragma unroll
        for (int r = 0; r < 8; ++r)
            out[(size_t)(row0 + wave * 16 + r + 8 * hl) * EMBED
                + col0 + t * 16 + l16] = acc[t][r];
}

// ---------------------------------------------------------------------------
// Host launcher
// ---------------------------------------------------------------------------
extern "C" void kernel_launch(void* const* d_in, const int* in_sizes, int n_in,
                              void* d_out, int out_size, void* d_ws, size_t ws_size,
                              hipStream_t stream)
{
    const float* x  = (const float*)d_in[0];
    const float* Wq = (const float*)d_in[1];
    const float* bq = (const float*)d_in[2];
    const float* Wk = (const float*)d_in[3];
    const float* bk = (const float*)d_in[4];
    const float* Wv = (const float*)d_in[5];
    const float* bv = (const float*)d_in[6];
    const float* Wo = (const float*)d_in[7];
    const float* bo = (const float*)d_in[8];
    float* out = (float*)d_out;

    // workspace layout (bf16): Q | K | V | H, each ROWS*DH elements (4 MB)
    const size_t seg = (size_t)ROWS * DH * sizeof(__bf16);
    char* ws = (char*)d_ws;
    __bf16* Qbf = (__bf16*)(ws + 0 * seg);
    __bf16* Kbf = (__bf16*)(ws + 1 * seg);
    __bf16* Vbf = (__bf16*)(ws + 2 * seg);
    __bf16* Hbf = (__bf16*)(ws + 3 * seg);

    const float qscale = 0.08838834764831845f;   // 1/sqrt(128)

    dim3 blkA(256), grdA(ROWS / 128);
    hipLaunchKernelGGL(qkv_proj_kernel, grdA, blkA, 0, stream, x, Wq, bq, Qbf, qscale);
    hipLaunchKernelGGL(qkv_proj_kernel, grdA, blkA, 0, stream, x, Wk, bk, Kbf, 1.0f);
    hipLaunchKernelGGL(qkv_proj_kernel, grdA, blkA, 0, stream, x, Wv, bv, Vbf, 1.0f);

    dim3 blkB(128), grdB(SEQ / 64, BATCH);
    hipLaunchKernelGGL(attn_kernel, grdB, blkB, 0, stream, Qbf, Kbf, Vbf, Hbf);

    dim3 blkC(256), grdC(ROWS / 128, EMBED / 128);
    hipLaunchKernelGGL(out_proj_kernel, grdC, blkC, 0, stream, Hbf, Wo, bo, out);
}
